// AtomEncoder_16003048144908
// MI455X (gfx1250) — matrix operand order repaired and checked by reading, not compiled, not run on hardware
//
#include <hip/hip_runtime.h>
#include <hip/hip_bf16.h>
#include <stdint.h>

typedef __bf16 bf16;
typedef float v8f  __attribute__((ext_vector_type(8)));
typedef bf16  v16bf __attribute__((ext_vector_type(16)));

#define B_   128
#define L_   192
#define DIM_ 512
#define M_   (B_ * L_)   // 24576 tokens
#define HID_ 2048

// ---------------------------------------------------------------------------
// Embedding: emb[l*B+b, :] = E_elem[element[b,l]] + pe[l] + E_aroma[aroma]
//                          + E_charge[charge+6] + E_seg[segment]
// Writes f32 (residual chain / final add) and bf16 (GEMM A input).
// ---------------------------------------------------------------------------
__global__ __launch_bounds__(128)
void embed_kernel(const int* __restrict__ elem, const int* __restrict__ aroma,
                  const int* __restrict__ charge, const int* __restrict__ seg,
                  const float* __restrict__ pe,
                  const float* __restrict__ Ee, const float* __restrict__ Ec,
                  const float* __restrict__ Ea, const float* __restrict__ Es,
                  float* __restrict__ emb, bf16* __restrict__ embb) {
  const int row = blockIdx.x;        // row = l*B + b
  const int l = row / B_;
  const int b = row % B_;
  const int d = threadIdx.x * 4;     // 128 threads * 4 floats = 512

  const int e  = elem[b * L_ + l];
  const int ar = aroma[b * L_ + l];
  const int ch = charge[b * L_ + l] + 6;
  const int sg = seg[b * L_ + l];

  float4 v  = *(const float4*)(Ee + (size_t)e  * DIM_ + d);
  float4 vp = *(const float4*)(pe + (size_t)l  * DIM_ + d);
  float4 va = *(const float4*)(Ea + (size_t)ar * DIM_ + d);
  float4 vc = *(const float4*)(Ec + (size_t)ch * DIM_ + d);
  float4 vs = *(const float4*)(Es + (size_t)sg * DIM_ + d);
  v.x += vp.x + va.x + vc.x + vs.x;
  v.y += vp.y + va.y + vc.y + vs.y;
  v.z += vp.z + va.z + vc.z + vs.z;
  v.w += vp.w + va.w + vc.w + vs.w;

  *(float4*)(emb + (size_t)row * DIM_ + d) = v;

  union { uint2 u; bf16 h[4]; } pk;
  pk.h[0] = (bf16)v.x; pk.h[1] = (bf16)v.y;
  pk.h[2] = (bf16)v.z; pk.h[3] = (bf16)v.w;
  *(uint2*)(embb + (size_t)row * DIM_ + d) = pk.u;
}

// ---------------------------------------------------------------------------
// Weight convert + transpose: Wt[n*K + k] = (bf16)W[k*N + n]
// Transposed layout makes the WMMA B-fragment two K-contiguous 16B loads per lane.
// ---------------------------------------------------------------------------
__global__ __launch_bounds__(256)
void wconvert_kernel(const float* __restrict__ W, bf16* __restrict__ Wt,
                     int K, int N) {
  int idx = blockIdx.x * blockDim.x + threadIdx.x;
  if (idx >= K * N) return;
  int k = idx / N;
  int n = idx % N;
  Wt[(size_t)n * K + k] = (bf16)W[idx];
}

// ---------------------------------------------------------------------------
// WMMA GEMM: out = act(A[M,K] @ W[K,N] + bias) (+ residual)
//   A  : bf16 row-major [M, K]
//   Bt : bf16 [N, K]   (pre-transposed weights)
// Wave tile 32x64: 2(M) x 4(N) v_wmma_f32_16x16x32_bf16 accumulators
//   -> 8 WMMAs per 12 fragment loads per K=32 step, A fragments reused 4x.
// Block = 8 waves arranged 4(M) x 2(N) -> 128x128 block tile.
// A fragment per lane (ISA 16-bit A 16x32 layout):
//   lanes 0-15 : row m0+lane,    K = {k0..k0+7, k0+16..k0+23}
//   lanes16-31 : row m0+lane-16, K = {k0+8..k0+15, k0+24..k0+31}
// B fragment per lane (ISA B 32x16 layout):
//   lanes 0-15 : col n0+lane,    K = {k0..k0+7, k0+16..k0+23}
//   lanes16-31 : col n0+lane-16, K = {k0+8..k0+15, k0+24..k0+31}
// C/D: lane -> N column, VGPR r -> M row r (lo half) / 8+r (hi half).
// ---------------------------------------------------------------------------
template <bool RELU, bool RES, bool OUTF, bool OUTB>
__global__ __launch_bounds__(256)
void gemm_wmma_bf16(const bf16* __restrict__ A, const bf16* __restrict__ Bt,
                    const float* __restrict__ bias, const float* __restrict__ res,
                    float* __restrict__ outF, bf16* __restrict__ outB,
                    int K, int N) {
  const int lane  = threadIdx.x & 31;
  const int wave  = threadIdx.x >> 5;
  const int waveM = wave >> 1;                 // 0..3
  const int waveN = wave & 1;                  // 0..1
  const int m0 = blockIdx.x * 128 + waveM * 32;
  const int n0 = blockIdx.y * 128 + waveN * 64;
  const int lh = lane & 15;
  const int hi = lane >> 4;

  const bf16* aptr0 = A + (size_t)(m0 + lh) * K + hi * 8;    // M tile 0
  const bf16* aptr1 = aptr0 + (size_t)16 * K;                // M tile 1
  const bf16* bptr[4];
#pragma unroll
  for (int j = 0; j < 4; ++j)
    bptr[j] = Bt + (size_t)(n0 + j * 16 + lh) * K + hi * 8;

  v8f acc[2][4] = {};

  for (int k0 = 0; k0 < K; k0 += 32) {
    union { v16bf v; uint4 q[2]; } a0, a1, bf[4];
    a0.q[0] = *(const uint4*)(aptr0 + k0);
    a0.q[1] = *(const uint4*)(aptr0 + k0 + 16);
    a1.q[0] = *(const uint4*)(aptr1 + k0);
    a1.q[1] = *(const uint4*)(aptr1 + k0 + 16);
#pragma unroll
    for (int j = 0; j < 4; ++j) {
      bf[j].q[0] = *(const uint4*)(bptr[j] + k0);
      bf[j].q[1] = *(const uint4*)(bptr[j] + k0 + 16);
    }

    if (k0 + 32 < K) {   // pull next K-slab toward WGP caches (locality 3)
      __builtin_prefetch(aptr0 + k0 + 32, 0, 3);
      __builtin_prefetch(bptr[0] + k0 + 32, 0, 3);
      __builtin_prefetch(bptr[2] + k0 + 32, 0, 3);
    }

#pragma unroll
    for (int j = 0; j < 4; ++j) {
      acc[0][j] = __builtin_amdgcn_wmma_f32_16x16x32_bf16(
          false, a0.v, false, bf[j].v, (short)0, acc[0][j], false, false);
      acc[1][j] = __builtin_amdgcn_wmma_f32_16x16x32_bf16(
          false, a1.v, false, bf[j].v, (short)0, acc[1][j], false, false);
    }
  }

#pragma unroll
  for (int i = 0; i < 2; ++i) {
#pragma unroll
    for (int j = 0; j < 4; ++j) {
      const int n  = n0 + j * 16 + lh;
      const int mb = m0 + i * 16 + hi * 8;
      const float bs = bias[n];
#pragma unroll
      for (int r = 0; r < 8; ++r) {
        size_t idx = (size_t)(mb + r) * N + n;
        float v = acc[i][j][r] + bs;
        if (RELU) v = fmaxf(v, 0.0f);
        if (RES)  v += res[idx];
        if (OUTF) outF[idx] = v;
        if (OUTB) outB[idx] = (bf16)v;
      }
    }
  }
}

// ---------------------------------------------------------------------------
// Bond aggregation: out[l*B+b,:] = emb[l*B+b,:]
//   + sum_{m: bond[b,l,m] != l} msg[bond[b,l,m]*B + b, :]
// ---------------------------------------------------------------------------
__global__ __launch_bounds__(128)
void agg_kernel(const float* __restrict__ emb, const float* __restrict__ msg,
                const int* __restrict__ bond, float* __restrict__ out) {
  const int bl = blockIdx.x;        // b*L + l
  const int b = bl / L_;
  const int l = bl % L_;
  const int orow = l * B_ + b;
  const int t = threadIdx.x;        // 128 threads * float4

  float4 s = *((const float4*)(emb + (size_t)orow * DIM_) + t);
  const int* bd = bond + (size_t)bl * 6;
#pragma unroll
  for (int m = 0; m < 6; ++m) {
    int l2 = bd[m];                 // uniform per block -> no divergence
    if (l2 != l) {
      float4 g = *((const float4*)(msg + (size_t)(l2 * B_ + b) * DIM_) + t);
      s.x += g.x; s.y += g.y; s.z += g.z; s.w += g.w;
    }
  }
  *((float4*)(out + (size_t)orow * DIM_) + t) = s;
}

// ---------------------------------------------------------------------------
extern "C" void kernel_launch(void* const* d_in, const int* in_sizes, int n_in,
                              void* d_out, int out_size, void* d_ws, size_t ws_size,
                              hipStream_t stream) {
  const int*   element = (const int*)d_in[0];
  const int*   bond    = (const int*)d_in[1];
  const int*   aroma   = (const int*)d_in[2];
  const int*   charge  = (const int*)d_in[3];
  const int*   segment = (const int*)d_in[4];
  const float* pe      = (const float*)d_in[5];
  const float* Ee      = (const float*)d_in[6];
  const float* Ec      = (const float*)d_in[7];
  const float* Ea      = (const float*)d_in[8];
  const float* Es      = (const float*)d_in[9];
  const float* W1 = (const float*)d_in[10]; const float* b1 = (const float*)d_in[11];
  const float* W2 = (const float*)d_in[12]; const float* b2 = (const float*)d_in[13];
  const float* W3 = (const float*)d_in[14]; const float* b3 = (const float*)d_in[15];
  const float* W4 = (const float*)d_in[16]; const float* b4 = (const float*)d_in[17];
  const float* W5 = (const float*)d_in[18]; const float* b5 = (const float*)d_in[19];
  float* out = (float*)d_out;

  // Workspace carve-up (~300 MB, all fully written before read -> deterministic)
  char* ws = (char*)d_ws;
  size_t off = 0;
  auto carve = [&](size_t bytes) -> void* {
    void* p = ws + off;
    off = (off + bytes + 255) & ~(size_t)255;
    return p;
  };
  float* emb  = (float*)carve((size_t)M_ * DIM_ * 4);   // kept to the end
  float* msg  = (float*)carve((size_t)M_ * DIM_ * 4);
  float* x1f  = (float*)carve((size_t)M_ * DIM_ * 4);   // x1, then x2 in place
  bf16*  hB   = (bf16*) carve((size_t)M_ * HID_ * 2);   // hidden (bf16)
  bf16*  aB   = (bf16*) carve((size_t)M_ * DIM_ * 2);   // emb bf16, later x2 bf16
  bf16*  x1B  = (bf16*) carve((size_t)M_ * DIM_ * 2);
  bf16*  W1t  = (bf16*) carve((size_t)DIM_ * HID_ * 2);
  bf16*  W2t  = (bf16*) carve((size_t)HID_ * DIM_ * 2);
  bf16*  W3t  = (bf16*) carve((size_t)DIM_ * HID_ * 2);
  bf16*  W4t  = (bf16*) carve((size_t)HID_ * DIM_ * 2);
  bf16*  W5t  = (bf16*) carve((size_t)DIM_ * DIM_ * 2);
  (void)ws_size; (void)in_sizes; (void)n_in; (void)out_size;

  // 1) embeddings
  embed_kernel<<<M_, 128, 0, stream>>>(element, aroma, charge, segment, pe,
                                       Ee, Ec, Ea, Es, emb, aB);

  // 2) weight convert+transpose to bf16 [N,K]
  {
    int n1 = DIM_ * HID_, n5 = DIM_ * DIM_;
    wconvert_kernel<<<(n1 + 255) / 256, 256, 0, stream>>>(W1, W1t, DIM_, HID_);
    wconvert_kernel<<<(n1 + 255) / 256, 256, 0, stream>>>(W2, W2t, HID_, DIM_);
    wconvert_kernel<<<(n1 + 255) / 256, 256, 0, stream>>>(W3, W3t, DIM_, HID_);
    wconvert_kernel<<<(n1 + 255) / 256, 256, 0, stream>>>(W4, W4t, HID_, DIM_);
    wconvert_kernel<<<(n5 + 255) / 256, 256, 0, stream>>>(W5, W5t, DIM_, DIM_);
  }

  // 3) FFN block 1: h = relu(emb@W1+b1); x1 = emb + h@W2 + b2
  gemm_wmma_bf16<true, false, false, true>
      <<<dim3(M_ / 128, HID_ / 128), 256, 0, stream>>>(
          aB, W1t, b1, nullptr, nullptr, hB, DIM_, HID_);
  gemm_wmma_bf16<false, true, true, true>
      <<<dim3(M_ / 128, DIM_ / 128), 256, 0, stream>>>(
          hB, W2t, b2, emb, x1f, x1B, HID_, DIM_);

  // 4) FFN block 2: h2 = relu(x1@W3+b3); x2 = x1 + h2@W4 + b4
  gemm_wmma_bf16<true, false, false, true>
      <<<dim3(M_ / 128, HID_ / 128), 256, 0, stream>>>(
          x1B, W3t, b3, nullptr, nullptr, hB, DIM_, HID_);
  gemm_wmma_bf16<false, true, true, true>
      <<<dim3(M_ / 128, DIM_ / 128), 256, 0, stream>>>(
          hB, W4t, b4, x1f, x1f, aB, HID_, DIM_);

  // 5) message = x2 @ W5 + b5
  gemm_wmma_bf16<false, false, true, false>
      <<<dim3(M_ / 128, DIM_ / 128), 256, 0, stream>>>(
          aB, W5t, b5, nullptr, msg, nullptr, DIM_, DIM_);

  // 6) bond gather + masked sum + emb
  agg_kernel<<<B_ * L_, 128, 0, stream>>>(emb, msg, bond, out);
}
